// AttentionOptimized_30562987278892
// MI455X (gfx1250) — matrix operand order repaired and checked by reading, compiled, not run on hardware
//
#include <hip/hip_runtime.h>
#include <hip/hip_bf16.h>

// ---- CDNA5 wave32 WMMA types -------------------------------------------------
typedef __attribute__((ext_vector_type(16))) __bf16 bf16x16;
typedef __attribute__((ext_vector_type(8)))  __bf16 bf16x8;
typedef __attribute__((ext_vector_type(8)))  float  f32x8;
typedef __attribute__((ext_vector_type(4)))  float  f32x4;

#define SEQ    8192
#define DIN    512
#define DHEAD  64
#define KBLOCK 64                 // keys per inner iteration
#define WAVES  4                  // waves per workgroup (split-K over keys)
#define KEYS_PER_WAVE (SEQ / WAVES)
#define LOG2E  1.44269504088896340736f

#if __has_builtin(__builtin_amdgcn_exp2f)
#define EXP2(x) __builtin_amdgcn_exp2f(x)
#else
#define EXP2(x) exp2f(x)
#endif

// =============================================================================
// Kernel 1: projections.  One wave computes a 16x64 tile of Q = (x@Wq)*log2e/8,
// K = x@Wk, or V = x@Wv with v_wmma_f32_16x16x32_bf16.
// Outputs: Q,K bf16 [SEQ][64]; V transposed bf16 [64][SEQ].
// Scores are produced directly in the exp2 domain (log2e folded into Q).
// =============================================================================
__global__ __launch_bounds__(32) void attn_proj_kernel(
    const float* __restrict__ x,
    const float* __restrict__ Wq,
    const float* __restrict__ Wk,
    const float* __restrict__ Wv,
    __bf16* __restrict__ Qb,
    __bf16* __restrict__ Kb,
    __bf16* __restrict__ Vt)
{
    const int lane  = threadIdx.x;      // 0..31 (wave32)
    const int n     = lane & 15;
    const int h     = lane >> 4;
    const int rbase = blockIdx.x * 16;
    const int which = blockIdx.y;       // 0=Q, 1=K, 2=V
    const float* W  = (which == 0) ? Wq : (which == 1) ? Wk : Wv;

    f32x8 acc[4];
    #pragma unroll
    for (int t = 0; t < 4; ++t) acc[t] = {};

    const int row = rbase + n;
    for (int c = 0; c < 16; ++c) {      // K-dim 512 in chunks of 32
        // A fragment (16x32 bf16): lane holds x[row][32c + 16h + i], i=0..15
        const float* xp = x + row * DIN + 32 * c + 16 * h;
        f32x4 a0 = *(const f32x4*)(xp);
        f32x4 a1 = *(const f32x4*)(xp + 4);
        f32x4 a2 = *(const f32x4*)(xp + 8);
        f32x4 a3 = *(const f32x4*)(xp + 12);
        bf16x16 afrag;
        #pragma unroll
        for (int i = 0; i < 4; ++i) {
            afrag[i]      = (__bf16)a0[i];
            afrag[4 + i]  = (__bf16)a1[i];
            afrag[8 + i]  = (__bf16)a2[i];
            afrag[12 + i] = (__bf16)a3[i];
        }
        #pragma unroll
        for (int t = 0; t < 4; ++t) {
            // B fragment (32x16 bf16): lane holds W[32c+16h+i][16t+n], i=0..15
            const float* wp = W + (32 * c + 16 * h) * DHEAD + 16 * t + n;
            bf16x16 bfrag;
            #pragma unroll
            for (int i = 0; i < 16; ++i) bfrag[i] = (__bf16)wp[i * DHEAD];
            acc[t] = __builtin_amdgcn_wmma_f32_16x16x32_bf16(
                false, afrag, false, bfrag, (short)0, acc[t], false, false);
        }
    }

    // C/D layout: acc[t][v] = out[m = v + 8h][n] of 16-wide tile t.
    if (which < 2) {
        __bf16* dst   = (which == 0) ? Qb : Kb;
        const float s = (which == 0) ? (0.125f * LOG2E) : 1.0f;
        #pragma unroll
        for (int t = 0; t < 4; ++t)
            #pragma unroll
            for (int v = 0; v < 8; ++v)
                dst[(rbase + v + 8 * h) * DHEAD + 16 * t + n] =
                    (__bf16)(acc[t][v] * s);
    } else {
        #pragma unroll
        for (int t = 0; t < 4; ++t)
            #pragma unroll
            for (int v = 0; v < 8; ++v)
                Vt[(16 * t + n) * SEQ + rbase + v + 8 * h] = (__bf16)acc[t][v];
    }
}

// =============================================================================
// Kernel 2: flash attention, split-K.  4 waves per workgroup; each wave owns
// the same 16 query rows but a disjoint 2048-key range (online softmax in the
// exp2 domain), 64 keys / iteration, 16 WMMAs per iteration.  Partials
// (m, l, O) merged through LDS by wave 0.
// =============================================================================
__global__ __launch_bounds__(32 * WAVES) void attn_flash_kernel(
    const __bf16* __restrict__ Qb,
    const __bf16* __restrict__ Kb,
    const __bf16* __restrict__ Vt,
    float* __restrict__ out)
{
    __shared__ __align__(16) __bf16 Plds[WAVES][16 * KBLOCK];  // per-wave P staging
    __shared__ __align__(16) float  Of[WAVES - 1][16][DHEAD];  // partial O tiles
    __shared__ float Sm[WAVES - 1][16];                        // partial row max
    __shared__ float Sl[WAVES - 1][16];                        // partial row sum

    const int tid   = threadIdx.x;
    const int w     = tid >> 5;        // wave id 0..3
    const int lane  = tid & 31;
    const int n     = lane & 15;
    const int h     = lane >> 4;
    const int qbase = blockIdx.x * 16;

    // Q A-fragments (K-chunks 0..31 / 32..63) — resident for the whole kernel.
    bf16x16 qa[2];
    {
        const __bf16* qp = Qb + (qbase + n) * DHEAD;
        #pragma unroll
        for (int c = 0; c < 2; ++c) {
            bf16x8 lo = *(const bf16x8*)(qp + 32 * c + 16 * h);
            bf16x8 hi = *(const bf16x8*)(qp + 32 * c + 16 * h + 8);
            #pragma unroll
            for (int i = 0; i < 8; ++i) { qa[c][i] = lo[i]; qa[c][8 + i] = hi[i]; }
        }
    }

    f32x8 O[4];
    #pragma unroll
    for (int t = 0; t < 4; ++t) O[t] = {};
    float mrun[8], lrun[8];
    #pragma unroll
    for (int v = 0; v < 8; ++v) { mrun[v] = -__builtin_inff(); lrun[v] = 0.0f; }

    __bf16* pl = &Plds[w][0];
    const int kbeg = w * KEYS_PER_WAVE;

    for (int jb = kbeg; jb < kbeg + KEYS_PER_WAVE; jb += KBLOCK) {
        // software prefetch of next key/value block (L2-resident, warm L0/L1)
        if (jb + KBLOCK < kbeg + KEYS_PER_WAVE) {
            __builtin_prefetch(Kb + (jb + KBLOCK + lane) * DHEAD, 0, 3);
            __builtin_prefetch(Kb + (jb + KBLOCK + 32 + lane) * DHEAD, 0, 3);
            __builtin_prefetch(Vt + lane * SEQ + jb + KBLOCK, 0, 3);
            __builtin_prefetch(Vt + (32 + lane) * SEQ + jb + KBLOCK, 0, 3);
        }

        // ---- S = Q @ K^T : four 16x16 score tiles over this 64-key block ----
        f32x8 S[4];
        #pragma unroll
        for (int s = 0; s < 4; ++s) {
            f32x8 sacc = {};
            const __bf16* kp = Kb + (jb + 16 * s + n) * DHEAD + 16 * h;
            #pragma unroll
            for (int c = 0; c < 2; ++c) {
                bf16x8 b0 = *(const bf16x8*)(kp + 32 * c);
                bf16x8 b1 = *(const bf16x8*)(kp + 32 * c + 8);
                bf16x16 bfrag;
                #pragma unroll
                for (int i = 0; i < 8; ++i) { bfrag[i] = b0[i]; bfrag[8 + i] = b1[i]; }
                sacc = __builtin_amdgcn_wmma_f32_16x16x32_bf16(
                    false, qa[c], false, bfrag, (short)0, sacc, false, false);
            }
            S[s] = sacc;
        }

        // ---- online softmax in exp2 domain (row m = v + 8h across half-wave) ----
        #pragma unroll
        for (int v = 0; v < 8; ++v) {
            float e0 = S[0][v], e1 = S[1][v], e2 = S[2][v], e3 = S[3][v];
            float rmax = fmaxf(fmaxf(e0, e1), fmaxf(e2, e3));
            #pragma unroll
            for (int mk = 1; mk < 16; mk <<= 1)
                rmax = fmaxf(rmax, __shfl_xor(rmax, mk, 32));
            float newm = fmaxf(mrun[v], rmax);
            float corr = EXP2(mrun[v] - newm);
            mrun[v] = newm;
            float p0 = EXP2(e0 - newm);
            float p1 = EXP2(e1 - newm);
            float p2 = EXP2(e2 - newm);
            float p3 = EXP2(e3 - newm);
            float rsum = (p0 + p1) + (p2 + p3);
            #pragma unroll
            for (int mk = 1; mk < 16; mk <<= 1)
                rsum += __shfl_xor(rsum, mk, 32);
            lrun[v] = lrun[v] * corr + rsum;
            #pragma unroll
            for (int t = 0; t < 4; ++t) O[t][v] *= corr;
            const int prow = (v + 8 * h) * KBLOCK + n;
            pl[prow]      = (__bf16)p0;
            pl[prow + 16] = (__bf16)p1;
            pl[prow + 32] = (__bf16)p2;
            pl[prow + 48] = (__bf16)p3;
        }
        asm volatile("s_wait_dscnt 0x0" ::: "memory");  // per-wave LDS store->load

        // reload P as two 16x32 A-fragments
        bf16x16 pa[2];
        {
            const __bf16* pp = pl + n * KBLOCK;
            #pragma unroll
            for (int cc = 0; cc < 2; ++cc) {
                bf16x8 lo = *(const bf16x8*)(pp + 32 * cc + 16 * h);
                bf16x8 hi = *(const bf16x8*)(pp + 32 * cc + 16 * h + 8);
                #pragma unroll
                for (int i = 0; i < 8; ++i) { pa[cc][i] = lo[i]; pa[cc][8 + i] = hi[i]; }
            }
        }
        asm volatile("" ::: "memory");  // keep loads before next iter's stores

        // ---- O += P @ V : B fragments are contiguous rows of V^T ----
        #pragma unroll
        for (int t = 0; t < 4; ++t) {
            #pragma unroll
            for (int cc = 0; cc < 2; ++cc) {
                const __bf16* vp = Vt + (16 * t + n) * SEQ + jb + 32 * cc + 16 * h;
                bf16x8 b0 = *(const bf16x8*)(vp);
                bf16x8 b1 = *(const bf16x8*)(vp + 8);
                bf16x16 vfrag;
                #pragma unroll
                for (int i = 0; i < 8; ++i) { vfrag[i] = b0[i]; vfrag[8 + i] = b1[i]; }
                O[t] = __builtin_amdgcn_wmma_f32_16x16x32_bf16(
                    false, pa[cc], false, vfrag, (short)0, O[t], false, false);
            }
        }
    }

    // ---- cross-wave merge of online-softmax partials ----
    if (w > 0) {
        #pragma unroll
        for (int t = 0; t < 4; ++t)
            #pragma unroll
            for (int v = 0; v < 8; ++v)
                Of[w - 1][v + 8 * h][16 * t + n] = O[t][v];
        if (n == 0) {
            #pragma unroll
            for (int v = 0; v < 8; ++v) {
                Sm[w - 1][v + 8 * h] = mrun[v];
                Sl[w - 1][v + 8 * h] = lrun[v];
            }
        }
    }
    __syncthreads();

    if (w == 0) {
        #pragma unroll
        for (int v = 0; v < 8; ++v) {
            const int m = v + 8 * h;
            float mf = mrun[v];
            #pragma unroll
            for (int ww = 0; ww < WAVES - 1; ++ww) mf = fmaxf(mf, Sm[ww][m]);
            float c0  = EXP2(mrun[v] - mf);
            float l   = lrun[v] * c0;
            float acc[4];
            #pragma unroll
            for (int t = 0; t < 4; ++t) acc[t] = O[t][v] * c0;
            #pragma unroll
            for (int ww = 0; ww < WAVES - 1; ++ww) {
                float cw = EXP2(Sm[ww][m] - mf);
                l += Sl[ww][m] * cw;
                #pragma unroll
                for (int t = 0; t < 4; ++t) acc[t] += Of[ww][m][16 * t + n] * cw;
            }
            float inv = 1.0f / l;
            #pragma unroll
            for (int t = 0; t < 4; ++t)
                out[(qbase + m) * DHEAD + 16 * t + n] = acc[t] * inv;
        }
    }
}

// =============================================================================
extern "C" void kernel_launch(void* const* d_in, const int* in_sizes, int n_in,
                              void* d_out, int out_size, void* d_ws, size_t ws_size,
                              hipStream_t stream)
{
    const float* x  = (const float*)d_in[0];
    const float* Wq = (const float*)d_in[1];
    const float* Wk = (const float*)d_in[2];
    const float* Wv = (const float*)d_in[3];

    __bf16* Qb = (__bf16*)d_ws;                    // [SEQ][64] bf16, pre-scaled
    __bf16* Kb = Qb + (size_t)SEQ * DHEAD;         // [SEQ][64] bf16
    __bf16* Vt = Kb + (size_t)SEQ * DHEAD;         // [64][SEQ] bf16 (transposed)

    attn_proj_kernel<<<dim3(SEQ / 16, 3), 32, 0, stream>>>(x, Wq, Wk, Wv, Qb, Kb, Vt);
    attn_flash_kernel<<<dim3(SEQ / 16), 32 * WAVES, 0, stream>>>(Qb, Kb, Vt, (float*)d_out);
}
